// GNNModel_54485955117135
// MI455X (gfx1250) — matrix operand order repaired and checked by reading, not compiled, not run on hardware
//
#include <hip/hip_runtime.h>
#include <hip/hip_bf16.h>

// MI455X / gfx1250 GNN pipeline. All dense GEMMs run on V_WMMA_F32_16X16X32_F16
// (f16 A/B, f32 accumulate). One wave owns a 16-row tile (16 edges or 16 nodes).
// Weights are pre-swizzled into per-(ktile,ntile,lane) 32B fragment chunks so a
// B fragment is one aligned v16h load (2x global_load_b128). Node features that
// only feed WMMA (xbn, x1, x2) are stored f16 so A fragments are pure b128 loads.

typedef __attribute__((ext_vector_type(16))) _Float16 v16h;
typedef __attribute__((ext_vector_type(8)))  _Float16 v8h;
typedef __attribute__((ext_vector_type(8)))  float    v8f;
typedef __attribute__((ext_vector_type(4)))  float    v4f;

#define WMMA_F16(a, b, c) \
  __builtin_amdgcn_wmma_f32_16x16x32_f16(false, (a), false, (b), (short)0, (c), false, false)

#define SHUF16(lo, hi) \
  __builtin_shufflevector((lo), (hi), 0, 1, 2, 3, 4, 5, 6, 7, 8, 9, 10, 11, 12, 13, 14, 15)

static __device__ __forceinline__ v8f vzero8() { v8f z = {}; return z; }

// A-matrix 16x32 f16 fragment layout (ISA 7.12.2): lane L: row M = L&15,
// K-half = (L>>4)*8; the 16 elements are two contiguous runs of 8:
//   k = k0 + kh + {0..7}   and   k = k0 + kh + 16 + {0..7}
static __device__ __forceinline__ int a_frag_k(int i, int lane) {
  int v = i >> 1;
  return ((v & 4) ? 16 : 0) + ((v & 3) << 1) + (i & 1) + ((lane >> 4) << 3);
}

// Pre-swizzled B fragment: tile t = kt*NT + nt holds 32 lanes x 16 f16 (32B/lane).
static __device__ __forceinline__ v16h load_b_sw(const _Float16* __restrict__ Wsw,
                                                 int NT, int kt, int nt, int lane) {
  return *(const v16h*)(Wsw + (((size_t)(kt * NT + nt)) << 9) + ((size_t)lane << 4));
}

// A fragment from a row-major f16 LDS tile (lda multiple of 8, base 16B aligned).
static __device__ __forceinline__ v16h load_a_lds(const _Float16* T, int lda,
                                                  int k0, int lane) {
  const _Float16* p = T + (lane & 15) * lda + k0 + ((lane >> 4) << 3);
  v8h lo = *(const v8h*)p;
  v8h hi = *(const v8h*)(p + 16);
  return SHUF16(lo, hi);
}

// A fragment from one global f16 row (row 16B aligned).
static __device__ __forceinline__ v16h load_a_gf16(const _Float16* __restrict__ row,
                                                   int k0, int lane) {
  const _Float16* p = row + k0 + ((lane >> 4) << 3);
  v8h lo = *(const v8h*)p;
  v8h hi = *(const v8h*)(p + 16);
  return SHUF16(lo, hi);
}

// One 8-run of the conv concat row [x[dst](DIN) | x[src](DIN) | e(128)], all f16.
// Segment boundaries are multiples of 32 -> each 8-run lies in one segment.
template <int DIN>
static __device__ __forceinline__ v8h run8h(const _Float16* __restrict__ xd,
                                            const _Float16* __restrict__ xs,
                                            const _Float16* __restrict__ er, int k) {
  if (k < DIN) return *(const v8h*)(xd + k);
  if (k < 2 * DIN) return *(const v8h*)(xs + (k - DIN));
  return *(const v8h*)(er + (k - 2 * DIN));
}

template <int DIN>
static __device__ __forceinline__ v16h load_a_concat(const _Float16* __restrict__ xd,
                                                     const _Float16* __restrict__ xs,
                                                     const _Float16* __restrict__ er,
                                                     int k0, int lane) {
  int kh = k0 + ((lane >> 4) << 3);
  v8h lo = run8h<DIN>(xd, xs, er, kh);
  v8h hi = run8h<DIN>(xd, xs, er, kh + 16);
  return SHUF16(lo, hi);
}

// ---------------------------------------------------------------- small kernels

// Swizzle W f32 [K,Nout] row-major -> f16 B-fragment chunks:
// D[(kt*NT+nt)*512 + lane*16 + i] = W[(kt*32 + (lane>>4)*8 + (i&7) + (i>>3)*16)*Nout + nt*16 + (lane&15)]
__global__ void swizzle_b_kernel(const float* __restrict__ W,
                                 _Float16* __restrict__ D, int K, int Nout) {
  int total = K * Nout;
  int idx = blockIdx.x * blockDim.x + threadIdx.x;
  if (idx >= total) return;
  int i    = idx & 15;
  int lane = (idx >> 4) & 31;
  int t    = idx >> 9;
  int ntn  = Nout >> 4;
  int kt = t / ntn, nt = t - kt * ntn;
  int n = nt * 16 + (lane & 15);
  int k = kt * 32 + ((lane >> 4) << 3) + (i & 7) + ((i >> 3) << 4);
  D[idx] = (_Float16)W[(size_t)k * Nout + n];
}

__global__ void bn_kernel(const float* __restrict__ x, const float* __restrict__ w,
                          const float* __restrict__ b, const float* __restrict__ rm,
                          const float* __restrict__ rv, _Float16* __restrict__ y,
                          long total) {
  long i = (long)blockIdx.x * blockDim.x + threadIdx.x;
  if (i < total) {
    int c = (int)(i & 63);
    y[i] = (_Float16)((x[i] - rm[c]) * rsqrtf(rv[c] + 1e-5f) * w[c] + b[c]);
  }
}

__global__ void deg_kernel(const int* __restrict__ eidx, float* __restrict__ deg,
                           int E) {
  int i = blockIdx.x * blockDim.x + threadIdx.x;
  if (i < E) atomicAdd(&deg[eidx[E + i]], 1.0f);
}

__global__ void invdeg_kernel(const float* __restrict__ deg,
                              float* __restrict__ inv, int N) {
  int i = blockIdx.x * blockDim.x + threadIdx.x;
  if (i < N) inv[i] = 1.0f / fmaxf(deg[i], 1.0f);
}

// ------------------------------------------------------------- edge encoder
__global__ __launch_bounds__(256) void edge_encoder_kernel(
    const float* __restrict__ edge_attr, const int* __restrict__ edge_index,
    const float* __restrict__ ln_g, const float* __restrict__ ln_b,
    const _Float16* __restrict__ W1, const float* __restrict__ B1,
    const _Float16* __restrict__ W2, const float* __restrict__ B2,
    const float* __restrict__ ecW1, const float* __restrict__ ecB1,
    const float* __restrict__ ecW2, const float* __restrict__ ecB2,
    _Float16* __restrict__ eOut, float* __restrict__ efsum, int E) {
  __shared__ __align__(16) _Float16 hT[8][16][72];
  int wid = threadIdx.x >> 5, lane = threadIdx.x & 31, m = lane & 15;
  long ebase = ((long)blockIdx.x * 8 + wid) * 16;
  long e = ebase + m;
  long ec = e < E ? e : (long)E - 1;
  int dstN = edge_index[(size_t)E + ec];

  const float* ea = edge_attr + ec * 32;
  float r[32], s0 = 0.f, s1 = 0.f;
#pragma unroll
  for (int k = 0; k < 32; k += 4) {
    v4f q = *(const v4f*)(ea + k);
#pragma unroll
    for (int j = 0; j < 4; ++j) { r[k + j] = q[j]; s0 += q[j]; s1 += q[j] * q[j]; }
  }
  float mu  = s0 * (1.f / 32.f);
  float var = s1 * (1.f / 32.f) - mu * mu;
  float rs  = rsqrtf(var + 1e-5f);

  // edge classifier gate (VALU; uniform weights become scalar loads)
  float acc2 = ecB2[0];
#pragma unroll
  for (int j = 0; j < 16; ++j) {
    float z = ecB1[j];
    for (int k = 0; k < 32; ++k) z += r[k] * ecW1[k * 16 + j];
    z = z > 0.f ? z : 0.f;
    acc2 += z * ecW2[j];
  }
  float ew = 1.f / (1.f + __expf(-acc2));

  // A fragment with LayerNorm applied inline (values already in registers)
  v16h a;
#pragma unroll
  for (int i = 0; i < 16; ++i) {
    int k = a_frag_k(i, lane);
    a[i] = (_Float16)(((r[k] - mu) * rs) * ln_g[k] + ln_b[k]);
  }

  // GEMM1: [16x32] x [32x64]
  v8f c0[4];
#pragma unroll
  for (int nt = 0; nt < 4; ++nt) c0[nt] = vzero8();
#pragma unroll
  for (int nt = 0; nt < 4; ++nt)
    c0[nt] = WMMA_F16(a, load_b_sw(W1, 4, 0, nt, lane), c0[nt]);

#pragma unroll
  for (int nt = 0; nt < 4; ++nt) {
    int n = nt * 16 + (lane & 15);
    float bb = B1[n];
#pragma unroll
    for (int rr = 0; rr < 8; ++rr) {
      int row = rr + ((lane >> 4) << 3);
      float v = c0[nt][rr] + bb;
      hT[wid][row][n] = (_Float16)(v > 0.f ? v : 0.f);
    }
  }
  __syncthreads();

  // GEMM2: [16x64] x [64x128]
  v8f c1[8];
#pragma unroll
  for (int nt = 0; nt < 8; ++nt) c1[nt] = vzero8();
#pragma unroll
  for (int kt = 0; kt < 2; ++kt) {
    v16h a2 = load_a_lds(&hT[wid][0][0], 72, kt * 32, lane);
#pragma unroll
    for (int nt = 0; nt < 8; ++nt)
      c1[nt] = WMMA_F16(a2, load_b_sw(W2, 8, kt, nt, lane), c1[nt]);
  }
#pragma unroll
  for (int nt = 0; nt < 8; ++nt) {
    int n = nt * 16 + (lane & 15);
    float bb = B2[n];
#pragma unroll
    for (int rr = 0; rr < 8; ++rr) {
      int row = rr + ((lane >> 4) << 3);
      long eg = ebase + row;
      int dn   = __shfl(dstN, row, 32);
      float we = __shfl(ew, row, 32);
      float v  = (c1[nt][rr] + bb) * we;
      if (eg < E) {
        eOut[eg * 128 + n] = (_Float16)v;
        atomicAdd(&efsum[(long)dn * 128 + n], v);
      }
    }
  }
}

// ------------------------------------------------------------- conv (message MLP)
template <int DIN, int DH, int DOUT>
__global__ __launch_bounds__(256) void conv_kernel(
    const _Float16* __restrict__ xn, const int* __restrict__ edge_index,
    const _Float16* __restrict__ eF,
    const _Float16* __restrict__ W1, const float* __restrict__ B1,
    const _Float16* __restrict__ W2, const float* __restrict__ B2,
    float* __restrict__ csum, int E) {
  constexpr int K1  = 2 * DIN + 128;
  constexpr int NT1 = DH / 16, KT1 = K1 / 32;
  constexpr int NT2 = DOUT / 16, KT2 = DH / 32;
  __shared__ __align__(16) _Float16 hT[8][16][DH + 8];

  int wid = threadIdx.x >> 5, lane = threadIdx.x & 31, m = lane & 15;
  long ebase = ((long)blockIdx.x * 8 + wid) * 16;
  long e = ebase + m;
  long ec = e < E ? e : (long)E - 1;
  int dn = edge_index[(size_t)E + ec];
  int sn = edge_index[ec];
  const _Float16* xd = xn + (long)dn * DIN;
  const _Float16* xs = xn + (long)sn * DIN;
  const _Float16* er = eF + ec * 128;

  v8f acc[NT1];
#pragma unroll
  for (int nt = 0; nt < NT1; ++nt) acc[nt] = vzero8();
#pragma unroll
  for (int kt = 0; kt < KT1; ++kt) {
    v16h a = load_a_concat<DIN>(xd, xs, er, kt * 32, lane);
#pragma unroll
    for (int nt = 0; nt < NT1; ++nt)
      acc[nt] = WMMA_F16(a, load_b_sw(W1, NT1, kt, nt, lane), acc[nt]);
  }
#pragma unroll
  for (int nt = 0; nt < NT1; ++nt) {
    int n = nt * 16 + (lane & 15);
    float bb = B1[n];
#pragma unroll
    for (int rr = 0; rr < 8; ++rr) {
      int row = rr + ((lane >> 4) << 3);
      float v = acc[nt][rr] + bb;
      hT[wid][row][n] = (_Float16)(v > 0.f ? v : 0.f);
    }
  }
  __syncthreads();

  v8f a2c[NT2];
#pragma unroll
  for (int nt = 0; nt < NT2; ++nt) a2c[nt] = vzero8();
#pragma unroll
  for (int kt = 0; kt < KT2; ++kt) {
    v16h a2 = load_a_lds(&hT[wid][0][0], DH + 8, kt * 32, lane);
#pragma unroll
    for (int nt = 0; nt < NT2; ++nt)
      a2c[nt] = WMMA_F16(a2, load_b_sw(W2, NT2, kt, nt, lane), a2c[nt]);
  }
#pragma unroll
  for (int nt = 0; nt < NT2; ++nt) {
    int n = nt * 16 + (lane & 15);
    float bb = B2[n];
#pragma unroll
    for (int rr = 0; rr < 8; ++rr) {
      int row = rr + ((lane >> 4) << 3);
      long eg = ebase + row;
      int dr  = __shfl(dn, row, 32);
      float v = a2c[nt][rr] + bb;
      if (eg < E) atomicAdd(&csum[(long)dr * DOUT + n], v);
    }
  }
}

// --------------------------------------------------- mean + LayerNorm + activation
// ACT=0 -> leaky_relu(0.01), ACT=1 -> relu.  OT: f16 for conv inputs, f32 for x3.
template <int D, int ACT, typename OT>
__global__ void finalize_kernel(const float* __restrict__ csum,
                                const float* __restrict__ invdeg,
                                const float* __restrict__ g,
                                const float* __restrict__ b,
                                OT* __restrict__ xout, int N) {
  constexpr int EPL = D / 32;
  int wid = threadIdx.x >> 5, lane = threadIdx.x & 31;
  long node = (long)blockIdx.x * (blockDim.x >> 5) + wid;
  if (node >= N) return;  // wave-uniform exit; no barriers below
  float id = invdeg[node];
  float v[EPL], s = 0.f;
#pragma unroll
  for (int j = 0; j < EPL; ++j) {
    v[j] = csum[node * D + lane + j * 32] * id;
    s += v[j];
  }
#pragma unroll
  for (int o = 16; o > 0; o >>= 1) s += __shfl_xor(s, o, 32);
  float mu = s / (float)D, q = 0.f;
#pragma unroll
  for (int j = 0; j < EPL; ++j) { float d0 = v[j] - mu; q += d0 * d0; }
#pragma unroll
  for (int o = 16; o > 0; o >>= 1) q += __shfl_xor(q, o, 32);
  float rsg = rsqrtf(q / (float)D + 1e-5f);
#pragma unroll
  for (int j = 0; j < EPL; ++j) {
    int c = lane + j * 32;
    float y = (v[j] - mu) * rsg * g[c] + b[c];
    y = ACT ? fmaxf(y, 0.f) : (y > 0.f ? y : 0.01f * y);
    xout[node * D + c] = (OT)y;
  }
}

// ---------------------------------------------------------------- node head
__global__ __launch_bounds__(128) void final_kernel(
    const _Float16* __restrict__ xbn, const float* __restrict__ x3,
    const float* __restrict__ efsum, const float* __restrict__ invdeg,
    const _Float16* __restrict__ projW, const float* __restrict__ projB,
    const _Float16* __restrict__ W1, const float* __restrict__ B1,
    const _Float16* __restrict__ W2, const float* __restrict__ B2,
    const float* __restrict__ W3, const float* __restrict__ B3,
    float* __restrict__ out, int N) {
  __shared__ __align__(16) _Float16 xfT[4][16][264];
  __shared__ __align__(16) _Float16 h1T[4][16][136];
  __shared__ __align__(16) _Float16 h2T[4][16][72];
  int wid = threadIdx.x >> 5, lane = threadIdx.x & 31, m = lane & 15;
  long nbase = ((long)blockIdx.x * 4 + wid) * 16;
  long node = nbase + m;
  long ncl = node < N ? node : (long)N - 1;
  const _Float16* xr = xbn + ncl * 64;

  // proj skip: [16x64] x [64x128]
  v8f acc[8];
#pragma unroll
  for (int nt = 0; nt < 8; ++nt) acc[nt] = vzero8();
#pragma unroll
  for (int kt = 0; kt < 2; ++kt) {
    v16h a = load_a_gf16(xr, kt * 32, lane);
#pragma unroll
    for (int nt = 0; nt < 8; ++nt)
      acc[nt] = WMMA_F16(a, load_b_sw(projW, 8, kt, nt, lane), acc[nt]);
  }
#pragma unroll
  for (int nt = 0; nt < 8; ++nt) {
    int n = nt * 16 + (lane & 15);
    float pb = projB[n];
#pragma unroll
    for (int rr = 0; rr < 8; ++rr) {
      int row = rr + ((lane >> 4) << 3);
      long nr = nbase + row;
      long nrc = nr < N ? nr : (long)N - 1;
      float v = 0.01f * (acc[nt][rr] + pb) + x3[nrc * 128 + n];
      xfT[wid][row][n] = (_Float16)v;
      if (nr < N) out[nr * 256 + n] = v;
    }
  }
  // edge-feature mean columns [128..256)
  for (int idx = lane; idx < 16 * 32; idx += 32) {
    int rr = idx >> 5, c4 = (idx & 31) << 2;
    long nr = nbase + rr;
    long nrc = nr < N ? nr : (long)N - 1;
    float iv = invdeg[nrc];
    v4f s4 = *(const v4f*)(efsum + nrc * 128 + c4);
#pragma unroll
    for (int j = 0; j < 4; ++j) {
      float v = s4[j] * iv;
      xfT[wid][rr][128 + c4 + j] = (_Float16)v;
      if (nr < N) out[nr * 256 + 128 + c4 + j] = v;
    }
  }
  __syncthreads();

  // h1 = elu(xf @ W1 + b1): [16x256] x [256x128]
  v8f a1[8];
#pragma unroll
  for (int nt = 0; nt < 8; ++nt) a1[nt] = vzero8();
#pragma unroll
  for (int kt = 0; kt < 8; ++kt) {
    v16h a = load_a_lds(&xfT[wid][0][0], 264, kt * 32, lane);
#pragma unroll
    for (int nt = 0; nt < 8; ++nt)
      a1[nt] = WMMA_F16(a, load_b_sw(W1, 8, kt, nt, lane), a1[nt]);
  }
#pragma unroll
  for (int nt = 0; nt < 8; ++nt) {
    int n = nt * 16 + (lane & 15);
    float bb = B1[n];
#pragma unroll
    for (int rr = 0; rr < 8; ++rr) {
      int row = rr + ((lane >> 4) << 3);
      float v = a1[nt][rr] + bb;
      h1T[wid][row][n] = (_Float16)(v > 0.f ? v : (__expf(v) - 1.f));
    }
  }
  __syncthreads();

  // h2 = elu(h1 @ W2 + b2): [16x128] x [128x64]
  v8f a2[4];
#pragma unroll
  for (int nt = 0; nt < 4; ++nt) a2[nt] = vzero8();
#pragma unroll
  for (int kt = 0; kt < 4; ++kt) {
    v16h a = load_a_lds(&h1T[wid][0][0], 136, kt * 32, lane);
#pragma unroll
    for (int nt = 0; nt < 4; ++nt)
      a2[nt] = WMMA_F16(a, load_b_sw(W2, 4, kt, nt, lane), a2[nt]);
  }
#pragma unroll
  for (int nt = 0; nt < 4; ++nt) {
    int n = nt * 16 + (lane & 15);
    float bb = B2[n];
#pragma unroll
    for (int rr = 0; rr < 8; ++rr) {
      int row = rr + ((lane >> 4) << 3);
      float v = a2[nt][rr] + bb;
      h2T[wid][row][n] = (_Float16)(v > 0.f ? v : (__expf(v) - 1.f));
    }
  }
  __syncthreads();

  // probs = h2 @ W3 + b3 (64 -> 1, VALU dot)
  if (lane < 16) {
    long nr = nbase + lane;
    if (nr < N) {
      float p = B3[0];
#pragma unroll
      for (int k = 0; k < 64; ++k) p += (float)h2T[wid][lane][k] * W3[k];
      out[(long)N * 256 + nr] = p;
    }
  }
}

// ---------------------------------------------------------------- launch

extern "C" void kernel_launch(void* const* d_in, const int* in_sizes, int n_in,
                              void* d_out, int out_size, void* d_ws, size_t ws_size,
                              hipStream_t stream) {
  (void)n_in; (void)out_size; (void)ws_size;
  const float* x_in  = (const float*)d_in[0];
  const int*   eidx  = (const int*)  d_in[1];
  const float* eattr = (const float*)d_in[2];
  const float* bn0w  = (const float*)d_in[3];
  const float* bn0b  = (const float*)d_in[4];
  const float* bn0rm = (const float*)d_in[5];
  const float* bn0rv = (const float*)d_in[6];
  const float* projw = (const float*)d_in[7];
  const float* projb = (const float*)d_in[8];
  const float* eelng = (const float*)d_in[9];
  const float* eelnb = (const float*)d_in[10];
  const float* eew1  = (const float*)d_in[11];
  const float* eeb1  = (const float*)d_in[12];
  const float* eew2  = (const float*)d_in[13];
  const float* eeb2  = (const float*)d_in[14];
  const float* ecw1  = (const float*)d_in[15];
  const float* ecb1  = (const float*)d_in[16];
  const float* ecw2  = (const float*)d_in[17];
  const float* ecb2  = (const float*)d_in[18];
  const float* c1w1  = (const float*)d_in[19];
  const float* c1b1  = (const float*)d_in[20];
  const float* c1w2  = (const float*)d_in[21];
  const float* c1b2  = (const float*)d_in[22];
  const float* c2w1  = (const float*)d_in[23];
  const float* c2b1  = (const float*)d_in[24];
  const float* c2w2  = (const float*)d_in[25];
  const float* c2b2  = (const float*)d_in[26];
  const float* c3w1  = (const float*)d_in[27];
  const float* c3b1  = (const float*)d_in[28];
  const float* c3w2  = (const float*)d_in[29];
  const float* c3b2  = (const float*)d_in[30];
  const float* ln1g  = (const float*)d_in[31];
  const float* ln1b  = (const float*)d_in[32];
  const float* ln2g  = (const float*)d_in[33];
  const float* ln2b  = (const float*)d_in[34];
  const float* ln3g  = (const float*)d_in[35];
  const float* ln3b  = (const float*)d_in[36];
  const float* npw1  = (const float*)d_in[37];
  const float* npb1  = (const float*)d_in[38];
  const float* npw2  = (const float*)d_in[39];
  const float* npb2  = (const float*)d_in[40];
  const float* npw3  = (const float*)d_in[41];
  const float* npb3  = (const float*)d_in[42];

  const int N = in_sizes[0] / 64;
  const int E = in_sizes[2] / 32;
  float* out = (float*)d_out;

  // workspace layout (deterministic, recomputed each call)
  char* ws = (char*)d_ws;
  size_t off = 0;
  auto take = [&](size_t bytes) -> char* {
    char* p = ws + off;
    off = (off + bytes + 255) & ~(size_t)255;
    return p;
  };
  _Float16* xbn   = (_Float16*)take((size_t)N * 64 * 2);
  float*    deg   = (float*)take((size_t)N * 4);
  float*    idg   = (float*)take((size_t)N * 4);
  _Float16* ef16  = (_Float16*)take((size_t)E * 128 * 2);
  float*    efsum = (float*)take((size_t)N * 128 * 4);
  float*    csum  = (float*)take((size_t)N * 128 * 4);
  _Float16* x1    = (_Float16*)take((size_t)N * 32 * 2);
  _Float16* x2    = (_Float16*)take((size_t)N * 64 * 2);
  float*    x3    = (float*)take((size_t)N * 128 * 4);
  _Float16* h_eew1 = (_Float16*)take(32 * 64 * 2);
  _Float16* h_eew2 = (_Float16*)take(64 * 128 * 2);
  _Float16* h_c1w1 = (_Float16*)take(256 * 32 * 2);
  _Float16* h_c1w2 = (_Float16*)take(32 * 32 * 2);
  _Float16* h_c2w1 = (_Float16*)take(192 * 64 * 2);
  _Float16* h_c2w2 = (_Float16*)take(64 * 64 * 2);
  _Float16* h_c3w1 = (_Float16*)take(256 * 128 * 2);
  _Float16* h_c3w2 = (_Float16*)take(128 * 128 * 2);
  _Float16* h_proj = (_Float16*)take(64 * 128 * 2);
  _Float16* h_np1  = (_Float16*)take(256 * 128 * 2);
  _Float16* h_np2  = (_Float16*)take(128 * 64 * 2);

  // pre-swizzle weights into B-fragment layout (tiny; L2-resident afterwards)
  auto sw = [&](const float* s, _Float16* d, int K, int Nout) {
    int n = K * Nout;
    swizzle_b_kernel<<<(n + 255) / 256, 256, 0, stream>>>(s, d, K, Nout);
  };
  sw(eew1, h_eew1, 32, 64);
  sw(eew2, h_eew2, 64, 128);
  sw(c1w1, h_c1w1, 256, 32);
  sw(c1w2, h_c1w2, 32, 32);
  sw(c2w1, h_c2w1, 192, 64);
  sw(c2w2, h_c2w2, 64, 64);
  sw(c3w1, h_c3w1, 256, 128);
  sw(c3w2, h_c3w2, 128, 128);
  sw(projw, h_proj, 64, 128);
  sw(npw1, h_np1, 256, 128);
  sw(npw2, h_np2, 128, 64);

  // bn0 (f16 output) + degree
  long tot = (long)N * 64;
  bn_kernel<<<(int)((tot + 255) / 256), 256, 0, stream>>>(x_in, bn0w, bn0b, bn0rm,
                                                          bn0rv, xbn, tot);
  hipMemsetAsync(deg, 0, (size_t)N * 4, stream);
  deg_kernel<<<(E + 255) / 256, 256, 0, stream>>>(eidx, deg, E);
  invdeg_kernel<<<(N + 255) / 256, 256, 0, stream>>>(deg, idg, N);

  // edge encoder (+ efsum accumulation)
  hipMemsetAsync(efsum, 0, (size_t)N * 128 * 4, stream);
  int eb = (E + 127) / 128;  // 8 waves x 16 edges per block
  edge_encoder_kernel<<<eb, 256, 0, stream>>>(eattr, eidx, eelng, eelnb, h_eew1,
                                              eeb1, h_eew2, eeb2, ecw1, ecb1, ecw2,
                                              ecb2, ef16, efsum, E);

  int nb8 = (N + 7) / 8;
  // conv1: DIN=64 -> 32
  hipMemsetAsync(csum, 0, (size_t)N * 32 * 4, stream);
  conv_kernel<64, 32, 32><<<eb, 256, 0, stream>>>(xbn, eidx, ef16, h_c1w1, c1b1,
                                                  h_c1w2, c1b2, csum, E);
  finalize_kernel<32, 0, _Float16><<<nb8, 256, 0, stream>>>(csum, idg, ln1g, ln1b,
                                                            x1, N);

  // conv2: DIN=32 -> 64
  hipMemsetAsync(csum, 0, (size_t)N * 64 * 4, stream);
  conv_kernel<32, 64, 64><<<eb, 256, 0, stream>>>(x1, eidx, ef16, h_c2w1, c2b1,
                                                  h_c2w2, c2b2, csum, E);
  finalize_kernel<64, 1, _Float16><<<nb8, 256, 0, stream>>>(csum, idg, ln2g, ln2b,
                                                            x2, N);

  // conv3: DIN=64 -> 128
  hipMemsetAsync(csum, 0, (size_t)N * 128 * 4, stream);
  conv_kernel<64, 128, 128><<<eb, 256, 0, stream>>>(x2, eidx, ef16, h_c3w1, c3b1,
                                                    h_c3w2, c3b2, csum, E);
  finalize_kernel<128, 1, float><<<nb8, 256, 0, stream>>>(csum, idg, ln3g, ln3b,
                                                          x3, N);

  // node head: xf + MLP
  final_kernel<<<(N + 63) / 64, 128, 0, stream>>>(xbn, x3, efsum, idg, h_proj,
                                                  projb, h_np1, npb1, h_np2, npb2,
                                                  npw3, npb3, out, N);
}
